// SamVisionLayer_67705864454284
// MI455X (gfx1250) — compile-verified, matrix operand order checked by reading, not run on hardware
//
#include <hip/hip_runtime.h>
#include <hip/hip_bf16.h>

typedef __bf16 bf16_t;
typedef __attribute__((ext_vector_type(16))) __bf16 v16bf;
typedef __attribute__((ext_vector_type(8)))  float  v8f;
typedef unsigned int v4u __attribute__((ext_vector_type(4)));
typedef int v8i __attribute__((ext_vector_type(8)));
typedef int v4i __attribute__((ext_vector_type(4)));

#define WMMA_BF16(a,b,c) \
  __builtin_amdgcn_wmma_f32_16x16x32_bf16(false,(a),false,(b),(short)0,(c),false,false)

#if __has_builtin(__builtin_amdgcn_tensor_load_to_lds) && __has_builtin(__builtin_amdgcn_s_wait_tensorcnt)
#define HAVE_TDM 1
#else
#define HAVE_TDM 0
#endif

union FragBF { v16bf v; bf16_t e[16]; };

__device__ __forceinline__ v8f zero8() {
  v8f z;
#pragma unroll
  for (int i = 0; i < 8; i++) z[i] = 0.f;
  return z;
}

// ---------------- constants ----------------
#define C_DIM 768
#define C3    2304
#define NHEAD 12
#define HD_   64
#define WSZ   14
#define NTOK  196      // WSZ*WSZ
#define TP    208      // token pad (13*16)
#define NWIN  200      // 8 images * 25 windows
#define MQ    41600    // NWIN*TP
#define NTOK2 32768    // B*H*W
#define MLPD  3072
#define SCALE_ 0.125f  // 64^-0.5

// ---------------- fp32 -> bf16 convert + transpose: src[K][N] -> dst[N][K] ---
__global__ void f2bf_t_kernel(const float* __restrict__ src, bf16_t* __restrict__ dst,
                              int Kd, int Nd) {
  long i = (long)blockIdx.x * 256 + threadIdx.x;
  if (i < (long)Kd * Nd) {
    int k = (int)(i / Nd), n = (int)(i % Nd);
    dst[(long)n * Kd + k] = (bf16_t)src[i];
  }
}

// ---------------- LayerNorm (row of 768) ----------------
__device__ __forceinline__ void ln_row(const float* __restrict__ src,
                                       const float* __restrict__ g,
                                       const float* __restrict__ b,
                                       bf16_t* __restrict__ dst, float* sbuf) {
  float s = 0.f, s2 = 0.f;
  for (int i = threadIdx.x; i < C_DIM; i += 128) { float v = src[i]; s += v; s2 += v * v; }
#pragma unroll
  for (int d = 16; d > 0; d >>= 1) { s += __shfl_xor(s, d, 32); s2 += __shfl_xor(s2, d, 32); }
  int wv = threadIdx.x >> 5, lane = threadIdx.x & 31;
  if (lane == 0) { sbuf[wv] = s; sbuf[4 + wv] = s2; }
  __syncthreads();
  s  = sbuf[0] + sbuf[1] + sbuf[2] + sbuf[3];
  s2 = sbuf[4] + sbuf[5] + sbuf[6] + sbuf[7];
  float mu = s * (1.f / C_DIM);
  float var = s2 * (1.f / C_DIM) - mu * mu;
  float rs = rsqrtf(var + 1e-6f);
  for (int i = threadIdx.x; i < C_DIM; i += 128)
    dst[i] = (bf16_t)((src[i] - mu) * rs * g[i] + b[i]);
}

// LN1 + pad + window partition -> Xw[NWIN][TP][C] bf16 (pad rows/positions = 0)
__global__ __launch_bounds__(128)
void ln_window_kernel(const float* __restrict__ x, const float* __restrict__ g,
                      const float* __restrict__ b, bf16_t* __restrict__ out) {
  __shared__ float sbuf[8];
  int tok = blockIdx.x;                 // 0..MQ-1
  int win = tok / TP, t = tok % TP;
  bf16_t* dst = out + (long)tok * C_DIM;
  int img = win / 25, wy = (win % 25) / 5, wx = win % 5;
  int y = wy * WSZ + t / WSZ, xx = wx * WSZ + t % WSZ;
  bool valid = (t < NTOK) && (y < 64) && (xx < 64);
  if (!valid) {
    for (int i = threadIdx.x; i < C_DIM; i += 128) dst[i] = (bf16_t)0.f;
    return;
  }
  const float* src = x + (((long)(img * 64 + y)) * 64 + xx) * C_DIM;
  ln_row(src, g, b, dst, sbuf);
}

// LN2 over h (in d_out) -> bf16
__global__ __launch_bounds__(128)
void ln2_kernel(const float* __restrict__ h, const float* __restrict__ g,
                const float* __restrict__ b, bf16_t* __restrict__ out) {
  __shared__ float sbuf[8];
  long tok = blockIdx.x;
  ln_row(h + tok * C_DIM, g, b, out + tok * C_DIM, sbuf);
}

// ---------------- TDM 2D tile load: 32 K-elems x 32 rows, stride K ----------
// LDS row stride is padded 64B + 16B via TDM pad feature -> 80B (40 elems).
#if HAVE_TDM
__device__ __forceinline__ void tdm_load_2d(const bf16_t* gsrc, unsigned lds_byte,
                                            int strideElems) {
  unsigned long long ga = (unsigned long long)(size_t)gsrc;
  v4u g0;
  g0[0] = 1u;                                              // count=1
  g0[1] = lds_byte;                                        // lds_addr
  g0[2] = (unsigned)ga;                                    // global_addr[31:0]
  g0[3] = (unsigned)((ga >> 32) & 0x01ffffffu) | (2u << 30); // ga[56:32] | type=2
  v8i g1;
  g1[0] = (int)((1u << 16) | (1u << 20) | (3u << 22) | (3u << 25));
  // data_size=2B | pad_enable | pad_interval: 16 DW (64B) | pad_amount: 4 DW (16B)
  const unsigned td = 0x40000000u;  // huge tensor dims -> no OOB clamping
  g1[1] = (int)((td & 0xffffu) << 16);                    // tensor_dim0[15:0]
  g1[2] = (int)((td >> 16) | ((td & 0xffffu) << 16));     // dim0[31:16] | dim1[15:0]
  g1[3] = (int)((td >> 16) | (32u << 16));                // dim1[31:16] | tile_dim0=32
  g1[4] = (int)32;                                        // tile_dim1=32, tile_dim2=0
  g1[5] = (int)(unsigned)strideElems;                     // tensor_dim0_stride[31:0]
  g1[6] = 0;                                              // stride[47:32] | dim1_stride lo
  g1[7] = 0;
  v4i gz = {0, 0, 0, 0};
#if defined(__clang_major__) && (__clang_major__ >= 23)
  v8i gz8 = {0, 0, 0, 0, 0, 0, 0, 0};
  __builtin_amdgcn_tensor_load_to_lds(g0, g1, gz, gz, gz8, 0);
#else
  __builtin_amdgcn_tensor_load_to_lds(g0, g1, gz, gz, 0);
#endif
}
#endif

// ---------------- WMMA GEMM: C[M,N] = A[M,K]@Bt[N,K]^T (+ fused epilogue) ----
// EPI 0: QKV scatter (V transposed)  1: proj+unwindow+residual  2: GELU->bf16  3: +=f32
struct GemmAux {
  bf16_t* q; bf16_t* k; bf16_t* v;
  float* outf; const float* resid;
  bf16_t* outbf;
};

template <int EPI>
__global__ __launch_bounds__(256)
void gemm128_kernel(const bf16_t* __restrict__ A, const bf16_t* __restrict__ Bt,
                    int K, const float* __restrict__ bias, GemmAux aux) {
  const int tid = threadIdx.x;
  const int wv = tid >> 5, lane = tid & 31;
  const int wm = wv >> 1, wn = wv & 1;
  const int hlf = lane >> 4, l16 = lane & 15;
  const int kbase = hlf * 8;
  const long rowBase = (long)blockIdx.y * 128;
  const int  colBase = blockIdx.x * 128;

  v8f acc[2][4];
#pragma unroll
  for (int i = 0; i < 2; i++)
#pragma unroll
    for (int j = 0; j < 4; j++) acc[i][j] = zero8();

#if HAVE_TDM
  __shared__ bf16_t smem[2][2][128 * 40];   // [buf][A|B][tile]
  const int ab = wv >> 2;                   // waves 0-3: A, 4-7: B
  const int qr = wv & 3;                    // 32-row quarter
  const bf16_t* gwave = ab ? (Bt + ((long)colBase + qr * 32) * K)
                           : (A + (rowBase + qr * 32) * K);
  const unsigned lds0 = (unsigned)(size_t)(void*)&smem[0][0][0];
  const unsigned waveOff = (unsigned)((ab * (128 * 40) + qr * (32 * 40)) * 2);
  const int nsteps = K >> 5;
  tdm_load_2d(gwave, lds0 + waveOff, K);
  for (int i = 0; i < nsteps; i++) {
    __builtin_amdgcn_s_wait_tensorcnt(0);
    __syncthreads();                        // tile i resident for all waves
    if (i + 1 < nsteps)
      tdm_load_2d(gwave + (i + 1) * 32,
                  lds0 + (unsigned)(((i + 1) & 1) * (2 * 128 * 40 * 2)) + waveOff, K);
    const bf16_t* sAc = &smem[i & 1][0][0];
    const bf16_t* sBc = &smem[i & 1][1][0];
    FragBF af[2], bfm[4];
#pragma unroll
    for (int ii = 0; ii < 2; ii++) {
      const bf16_t* s = sAc + (wm * 32 + ii * 16 + l16) * 40 + kbase;
#pragma unroll
      for (int e = 0; e < 8; e++) { af[ii].e[e] = s[e]; af[ii].e[8 + e] = s[16 + e]; }
    }
#pragma unroll
    for (int j = 0; j < 4; j++) {
      const bf16_t* s = sBc + (wn * 64 + j * 16 + l16) * 40 + kbase;
#pragma unroll
      for (int e = 0; e < 8; e++) { bfm[j].e[e] = s[e]; bfm[j].e[8 + e] = s[16 + e]; }
    }
#pragma unroll
    for (int ii = 0; ii < 2; ii++)
#pragma unroll
      for (int j = 0; j < 4; j++)
        acc[ii][j] = WMMA_BF16(af[ii].v, bfm[j].v, acc[ii][j]);
    __syncthreads();                        // done reading buf before it is re-filled
  }
#else
  __shared__ bf16_t sA[128 * 40];
  __shared__ bf16_t sB[128 * 40];
  const int ar = tid >> 1, aks = (tid & 1) * 16;
  for (int k0 = 0; k0 < K; k0 += 32) {
    const bf16_t* asrc = A + (rowBase + ar) * (long)K + k0 + aks;
    const bf16_t* bsrc = Bt + ((long)colBase + ar) * K + k0 + aks;
    bf16_t* adst = sA + ar * 40 + aks;
    bf16_t* bdst = sB + ar * 40 + aks;
#pragma unroll
    for (int i = 0; i < 16; i++) adst[i] = asrc[i];
#pragma unroll
    for (int i = 0; i < 16; i++) bdst[i] = bsrc[i];
    if (k0 + 32 < K) {
      __builtin_prefetch(asrc + 32, 0, 1);
      __builtin_prefetch(bsrc + 32, 0, 1);
    }
    __syncthreads();
    FragBF af[2], bfm[4];
#pragma unroll
    for (int ii = 0; ii < 2; ii++) {
      const bf16_t* s = sA + (wm * 32 + ii * 16 + l16) * 40 + kbase;
#pragma unroll
      for (int e = 0; e < 8; e++) { af[ii].e[e] = s[e]; af[ii].e[8 + e] = s[16 + e]; }
    }
#pragma unroll
    for (int j = 0; j < 4; j++) {
      const bf16_t* s = sB + (wn * 64 + j * 16 + l16) * 40 + kbase;
#pragma unroll
      for (int e = 0; e < 8; e++) { bfm[j].e[e] = s[e]; bfm[j].e[8 + e] = s[16 + e]; }
    }
#pragma unroll
    for (int ii = 0; ii < 2; ii++)
#pragma unroll
      for (int j = 0; j < 4; j++)
        acc[ii][j] = WMMA_BF16(af[ii].v, bfm[j].v, acc[ii][j]);
    __syncthreads();
  }
#endif

#pragma unroll
  for (int i = 0; i < 2; i++) {
#pragma unroll
    for (int jt = 0; jt < 4; jt++) {
#pragma unroll
      for (int j = 0; j < 8; j++) {
        long row = rowBase + wm * 32 + i * 16 + hlf * 8 + j;
        int col = colBase + wn * 64 + jt * 16 + l16;
        float val = acc[i][jt][j] + bias[col];
        if (EPI == 0) {  // split into Q/K head-major [bh][TP][64]; V -> [bh][64][TP]
          int win = (int)(row / TP), t = (int)(row % TP);
          int part = col / C_DIM, hh = (col % C_DIM) / HD_, d = col % HD_;
          long bh = (long)win * NHEAD + hh;
          if (part == 2)
            aux.v[(bh * HD_ + d) * TP + t] = (bf16_t)val;
          else
            (part == 0 ? aux.q : aux.k)[(bh * TP + t) * HD_ + d] = (bf16_t)val;
        } else if (EPI == 1) {  // proj + unwindow + residual
          int win = (int)(row / TP), t = (int)(row % TP);
          if (t < NTOK) {
            int img = win / 25, wy = (win % 25) / 5, wx = win % 5;
            int y = wy * WSZ + t / WSZ, xx = wx * WSZ + t % WSZ;
            if (y < 64 && xx < 64) {
              long idx = ((((long)img * 64) + y) * 64 + xx) * C_DIM + col;
              aux.outf[idx] = val + aux.resid[idx];
            }
          }
        } else if (EPI == 2) {  // exact GELU -> bf16
          float gl = 0.5f * val * (1.f + erff(val * 0.70710678118654752f));
          aux.outbf[row * MLPD + col] = (bf16_t)gl;
        } else {  // accumulate into f32 output
          long idx = row * C_DIM + col;
          aux.outf[idx] += val;
        }
      }
    }
  }
}

// ---------------- Attention: one block per (window, head) -------------------
__global__ __launch_bounds__(64)
void attn_kernel(const bf16_t* __restrict__ Q, const bf16_t* __restrict__ Km,
                 const bf16_t* __restrict__ Vt, const float* __restrict__ rph,
                 const float* __restrict__ rpw, bf16_t* __restrict__ out) {
  __shared__ bf16_t vt[64][224];        // V^T, zero-padded tokens
  __shared__ bf16_t pbuf[2][16][224];   // per-wave softmax rows for A-fragment
  __shared__ bf16_t relh[NTOK * WSZ];
  __shared__ bf16_t relw[NTOK * WSZ];

  const int bh = blockIdx.x;            // win*12 + head
  const int win = bh / NHEAD, head = bh % NHEAD;
  const long base = (long)bh * TP * HD_;
  const bf16_t* Qb = Q + base;
  const bf16_t* Kb = Km + base;
  const int tid = threadIdx.x;
  const int wv = tid >> 5, lane = tid & 31;
  const int hlf = lane >> 4, l16 = lane & 15;
  const int kbase = hlf * 8;

  // V^T already in memory: contiguous 16-element copies per d-row
  for (int idx = tid; idx < 64 * 13; idx += 64) {
    int d = idx / 13, seg = idx % 13;
    const bf16_t* src = Vt + ((long)bh * HD_ + d) * TP + seg * 16;
    bf16_t* dst = &vt[d][seg * 16];
#pragma unroll
    for (int i = 0; i < 16; i++) dst[i] = src[i];
  }
  {
    int d = tid;  // 64 threads == 64 d rows
#pragma unroll
    for (int i = 0; i < 16; i++) vt[d][208 + i] = (bf16_t)0.f;
  }
  for (int idx = tid; idx < NTOK * WSZ * 2; idx += 64) {
    int which = (idx >= NTOK * WSZ);
    int rem = which ? idx - NTOK * WSZ : idx;
    int qt = rem / WSZ, kc = rem % WSZ;
    int qy = qt / WSZ, qx = qt % WSZ;
    int ri = (which ? qx : qy) - kc + (WSZ - 1);
    const float* R = (which ? rpw : rph) + (long)ri * HD_;
    const bf16_t* qp = Qb + (long)qt * HD_;
    float s = 0.f;
#pragma unroll 8
    for (int d = 0; d < HD_; d++) s += (float)qp[d] * R[d];
    (which ? relw : relh)[rem] = (bf16_t)s;
  }
  __syncthreads();

  for (int m = wv; m < 13; m += 2) {
    FragBF qa[2];
    {
      const bf16_t* qp = Qb + (long)(m * 16 + l16) * HD_;
#pragma unroll
      for (int kk = 0; kk < 2; kk++) {
        const bf16_t* s = qp + kk * 32 + kbase;
#pragma unroll
        for (int e = 0; e < 8; e++) { qa[kk].e[e] = s[e]; qa[kk].e[8 + e] = s[16 + e]; }
      }
    }
    v8f sc[13];
#pragma unroll
    for (int n = 0; n < 13; n++) {
      v8f a = zero8();
      const bf16_t* kp = Kb + (long)(n * 16 + l16) * HD_;
#pragma unroll
      for (int kk = 0; kk < 2; kk++) {
        FragBF kf;
        const bf16_t* s = kp + kk * 32 + kbase;
#pragma unroll
        for (int e = 0; e < 8; e++) { kf.e[e] = s[e]; kf.e[8 + e] = s[16 + e]; }
        a = WMMA_BF16(qa[kk].v, kf.v, a);
      }
      int kt = n * 16 + l16;
      int ky = kt / WSZ, kx = kt % WSZ;
#pragma unroll
      for (int j = 0; j < 8; j++) {
        int qt = m * 16 + hlf * 8 + j;
        int qtc = qt < NTOK ? qt : NTOK - 1;
        float vsc;
        if (kt < NTOK)
          vsc = SCALE_ * a[j] + (float)relh[qtc * WSZ + ky] + (float)relw[qtc * WSZ + kx];
        else
          vsc = -1e30f;
        a[j] = vsc;
      }
      sc[n] = a;
    }
    // softmax along the 208-wide row (13 tiles x 16 lanes per half)
#pragma unroll
    for (int j = 0; j < 8; j++) {
      float mx = -1e30f;
#pragma unroll
      for (int n = 0; n < 13; n++) mx = fmaxf(mx, sc[n][j]);
#pragma unroll
      for (int d = 1; d < 16; d <<= 1) mx = fmaxf(mx, __shfl_xor(mx, d, 32));
      float sum = 0.f;
#pragma unroll
      for (int n = 0; n < 13; n++) { float p = __expf(sc[n][j] - mx); sc[n][j] = p; sum += p; }
#pragma unroll
      for (int d = 1; d < 16; d <<= 1) sum += __shfl_xor(sum, d, 32);
      float inv = 1.f / sum;
#pragma unroll
      for (int n = 0; n < 13; n++) sc[n][j] *= inv;
    }
    // stash P (wave-private LDS) so it can be read back in A-fragment layout
#pragma unroll
    for (int n = 0; n < 13; n++)
#pragma unroll
      for (int j = 0; j < 8; j++)
        pbuf[wv][hlf * 8 + j][n * 16 + l16] = (bf16_t)sc[n][j];
    for (int idx = lane; idx < 16 * 16; idx += 32)
      pbuf[wv][idx >> 4][208 + (idx & 15)] = (bf16_t)0.f;

    v8f oacc[4];
#pragma unroll
    for (int j = 0; j < 4; j++) oacc[j] = zero8();
#pragma unroll
    for (int ks = 0; ks < 7; ks++) {
      FragBF pf;
      const bf16_t* ps = &pbuf[wv][l16][ks * 32 + kbase];
#pragma unroll
      for (int e = 0; e < 8; e++) { pf.e[e] = ps[e]; pf.e[8 + e] = ps[16 + e]; }
#pragma unroll
      for (int dt = 0; dt < 4; dt++) {
        FragBF vf;
        const bf16_t* vs = &vt[dt * 16 + l16][ks * 32 + kbase];
#pragma unroll
        for (int e = 0; e < 8; e++) { vf.e[e] = vs[e]; vf.e[8 + e] = vs[16 + e]; }
        oacc[dt] = WMMA_BF16(pf.v, vf.v, oacc[dt]);
      }
    }
#pragma unroll
    for (int dt = 0; dt < 4; dt++)
#pragma unroll
      for (int j = 0; j < 8; j++) {
        int qt = m * 16 + hlf * 8 + j;
        int d = dt * 16 + l16;
        out[((long)win * TP + qt) * C_DIM + head * HD_ + d] = (bf16_t)oacc[dt][j];
      }
  }
}

// ---------------- host launcher ----------------
extern "C" void kernel_launch(void* const* d_in, const int* in_sizes, int n_in,
                              void* d_out, int out_size, void* d_ws, size_t ws_size,
                              hipStream_t stream) {
  const float* hs    = (const float*)d_in[0];
  const float* ln1g  = (const float*)d_in[1];
  const float* ln1b  = (const float*)d_in[2];
  const float* qkvw  = (const float*)d_in[3];
  const float* qkvb  = (const float*)d_in[4];
  const float* projw = (const float*)d_in[5];
  const float* projb = (const float*)d_in[6];
  const float* rph   = (const float*)d_in[7];
  const float* rpw   = (const float*)d_in[8];
  const float* ln2g  = (const float*)d_in[9];
  const float* ln2b  = (const float*)d_in[10];
  const float* lin1w = (const float*)d_in[11];
  const float* lin1b = (const float*)d_in[12];
  const float* lin2w = (const float*)d_in[13];
  const float* lin2b = (const float*)d_in[14];
  float* outp = (float*)d_out;
  char* ws = (char*)d_ws;

  const size_t SZ = (size_t)MQ * C_DIM * 2;  // 63.9 MB slot (also per-tensor QKV size)
  bf16_t* Xw   = (bf16_t*)(ws);              // slot 0
  bf16_t* attn = (bf16_t*)(ws);              // reuse slot 0 after QKV gemm
  bf16_t* ln2o = (bf16_t*)(ws);              // reuse slot 0 after proj
  bf16_t* Qb   = (bf16_t*)(ws + SZ);
  bf16_t* hid  = (bf16_t*)(ws + SZ);         // reuse Q slot after attention
  bf16_t* Kb   = (bf16_t*)(ws + 2 * SZ);
  bf16_t* Vtb  = (bf16_t*)(ws + 3 * SZ);     // V stored transposed [bh][64][208]
  bf16_t* qkvwt = (bf16_t*)(ws + 4 * SZ);    // all weights bf16, pre-TRANSPOSED [N][K]
  bf16_t* projwt = qkvwt + (size_t)C_DIM * C3;
  bf16_t* lin1wt = projwt + (size_t)C_DIM * C_DIM;
  bf16_t* lin2wt = lin1wt + (size_t)C_DIM * MLPD;

  auto cvtT = [&](const float* s, bf16_t* d, int Kd, int Nd) {
    long n = (long)Kd * Nd;
    f2bf_t_kernel<<<dim3((unsigned)((n + 255) / 256)), dim3(256), 0, stream>>>(s, d, Kd, Nd);
  };
  cvtT(qkvw, qkvwt, C_DIM, C3);
  cvtT(projw, projwt, C_DIM, C_DIM);
  cvtT(lin1w, lin1wt, C_DIM, MLPD);
  cvtT(lin2w, lin2wt, MLPD, C_DIM);

  // LN1 + window partition (padded windows = 0)
  ln_window_kernel<<<MQ, 128, 0, stream>>>(hs, ln1g, ln1b, Xw);

  // QKV projection, scattered into per-head Q/K and transposed V
  {
    GemmAux aux{}; aux.q = Qb; aux.k = Kb; aux.v = Vtb;
    gemm128_kernel<0><<<dim3(C3 / 128, MQ / 128), 256, 0, stream>>>(
        Xw, qkvwt, C_DIM, qkvb, aux);
  }

  // windowed attention with decomposed rel-pos
  attn_kernel<<<NWIN * NHEAD, 64, 0, stream>>>(Qb, Kb, Vtb, rph, rpw, attn);

  // output projection + unwindow + residual -> h in d_out
  {
    GemmAux aux{}; aux.outf = outp; aux.resid = hs;
    gemm128_kernel<1><<<dim3(C_DIM / 128, MQ / 128), 256, 0, stream>>>(
        attn, projwt, C_DIM, projb, aux);
  }

  // LN2 over h
  ln2_kernel<<<NTOK2, 128, 0, stream>>>(outp, ln2g, ln2b, ln2o);

  // MLP in 4 token chunks (bounds workspace): lin1+GELU -> bf16, lin2 += into h
  const int CHUNK = 8192;
  for (int c0 = 0; c0 < NTOK2; c0 += CHUNK) {
    GemmAux a1{}; a1.outbf = hid;
    gemm128_kernel<2><<<dim3(MLPD / 128, CHUNK / 128), 256, 0, stream>>>(
        ln2o + (long)c0 * C_DIM, lin1wt, C_DIM, lin1b, a1);
    GemmAux a2{}; a2.outf = outp + (long)c0 * C_DIM;
    gemm128_kernel<3><<<dim3(C_DIM / 128, CHUNK / 128), 256, 0, stream>>>(
        hid, lin2wt, MLPD, lin2b, a2);
  }
}
